// MultiHeadAttention_41695542510194
// MI455X (gfx1250) — matrix operand order repaired and checked by reading, not compiled, not run on hardware
//
#include <hip/hip_runtime.h>
#include <type_traits>

#define BB     2
#define SS     2048
#define DIN    1024
#define DMODEL 1024
#define NH     16
#define HD     64
#define MROWS  (BB * SS)   // 4096

typedef __attribute__((ext_vector_type(16))) __bf16 v16bf;
typedef __attribute__((ext_vector_type(8)))  __bf16 v8bf;
typedef __attribute__((ext_vector_type(8)))  float  v8f;

__device__ __forceinline__ __bf16 f2bf(float f) {
  unsigned u = __builtin_bit_cast(unsigned, f);
  u += 0x7fffu + ((u >> 16) & 1u);            // round-to-nearest-even
  unsigned short h = (unsigned short)(u >> 16);
  return __builtin_bit_cast(__bf16, h);
}

__device__ __forceinline__ v16bf cat16(v8bf lo, v8bf hi) {
  return __builtin_shufflevector(lo, hi, 0,1,2,3,4,5,6,7,8,9,10,11,12,13,14,15);
}

// ---- row-local (16-lane) reductions via DPP16 row_ror -- pure VALU, no LDS ----
template <int CTRL>
__device__ __forceinline__ float rowror(float x) {
  int i = __builtin_bit_cast(int, x);
  int r = __builtin_amdgcn_update_dpp(0, i, CTRL, 0xF, 0xF, true);
  return __builtin_bit_cast(float, r);
}
__device__ __forceinline__ float rowmax16(float m) {
  m = fmaxf(m, rowror<0x121>(m));   // row_ror:1
  m = fmaxf(m, rowror<0x122>(m));   // row_ror:2
  m = fmaxf(m, rowror<0x124>(m));   // row_ror:4
  m = fmaxf(m, rowror<0x128>(m));   // row_ror:8
  return m;
}
__device__ __forceinline__ float rowsum16(float s) {
  s += rowror<0x121>(s);
  s += rowror<0x122>(s);
  s += rowror<0x124>(s);
  s += rowror<0x128>(s);
  return s;
}

// gfx1250 async copy global -> LDS (ASYNCcnt), GVS addressing: saddr + voffset.
__device__ __forceinline__ void async_ld_b128(unsigned lds_byte_addr,
                                              unsigned voff_bytes,
                                              const void* sbase) {
  asm volatile("global_load_async_to_lds_b128 %0, %1, %2"
               :: "v"(lds_byte_addr), "v"(voff_bytes), "s"(sbase)
               : "memory");
}
__device__ __forceinline__ void wait_async0() {
  asm volatile("s_wait_asynccnt 0" ::: "memory");
}

// ---------------------------------------------------------------- cast x -> bf16
__global__ void cast_f32_bf16(const float* __restrict__ in,
                              __bf16* __restrict__ out, int n) {
  int i = (blockIdx.x * blockDim.x + threadIdx.x) * 4;
  if (i + 3 < n) {
    float4 v = *(const float4*)(in + i);
    out[i + 0] = f2bf(v.x);
    out[i + 1] = f2bf(v.y);
    out[i + 2] = f2bf(v.z);
    out[i + 3] = f2bf(v.w);
  }
}

// --------------------------------------------- transpose fp32 [R,C] -> bf16 [C,R]
__global__ void transpose_f32_bf16(const float* __restrict__ in,
                                   __bf16* __restrict__ out, int R, int C) {
  __shared__ float tile[32][33];
  int tx = threadIdx.x & 31, ty = threadIdx.x >> 5;   // 256 threads = 32x8
  int r0 = blockIdx.y * 32, c0 = blockIdx.x * 32;
#pragma unroll
  for (int i = 0; i < 4; ++i)
    tile[ty + i * 8][tx] = in[(size_t)(r0 + ty + i * 8) * C + (c0 + tx)];
  __syncthreads();
#pragma unroll
  for (int i = 0; i < 4; ++i)
    out[(size_t)(c0 + ty + i * 8) * R + (r0 + tx)] = f2bf(tile[tx][ty + i * 8]);
}

// ------------------------------------------------------------------ WMMA GEMM
// Block = 8 waves = 128x64 tile. B panel (64 rows x 32 k, bf16) async-staged
// into LDS (double-buffered); each wave owns 16 rows, reads B frags from LDS.
// A: [Mr,Kd] bf16 row-major.  Bt: [Nc,Kd] bf16 (B transposed).
// MODE 0: bf16 row-major store, scaled   (Q with 1/8*log2e, K with 1.0)
// MODE 1: bf16 per-head transposed store -> Vt [B,NH,HD,SS]
// MODE 2: f32 store + bias               (output projection)
template <int MODE>
__global__ __launch_bounds__(256) void gemm_bf16_wmma(
    const __bf16* __restrict__ A, const __bf16* __restrict__ Bt,
    void* __restrict__ Cout, const float* __restrict__ bias, float scale,
    int Mr, int Nc, int Kd) {
  __shared__ __align__(16) __bf16 bsh[2][64 * 32];   // 2 x 4KB B panels
  const int tid = threadIdx.x;
  const int lane = tid & 31, wave = tid >> 5;
  const int ln = lane & 15, half = lane >> 4;
  const int nT4 = Nc >> 6;
  const int bm = blockIdx.x / nT4;        // 128-row tile
  const int bn = blockIdx.x % nT4;        // 64-col tile (== head for MODE 1)
  const int tmRow = bm * 128 + wave * 16; // this wave's 16-row strip

  // async loader mapping: one b128 per thread per stage (64 rows x 32 k)
  const int lrow = tid >> 2;              // 0..63
  const int lofs = (tid & 3) * 8;         // bf16 offset within 32-k slice
  const unsigned ldsBase =
      (unsigned)(uintptr_t)(&bsh[0][0]) + (unsigned)(lrow * 64 + lofs * 2);
  const unsigned gBase = (unsigned)(((size_t)(bn * 64 + lrow) * Kd + lofs) * 2);

  async_ld_b128(ldsBase, gBase, Bt);      // prologue: stage k=0 into buffer 0
  wait_async0();
  __syncthreads();

  const __bf16* arow = A + (size_t)(tmRow + ln) * Kd;
  v8f acc[4] = {};
  int buf = 0;
  for (int k = 0; k < Kd; k += 32) {
    const bool hasNext = (k + 32) < Kd;
    if (hasNext) {
      async_ld_b128(ldsBase + (unsigned)((buf ^ 1) * 4096),
                    gBase + (unsigned)(k + 32) * 2, Bt);
      __builtin_prefetch(arow + k + 32, 0, 0);   // global_prefetch_b8 (A stream)
    }
    // A fragment: K in [half*16,+16)
    v16bf af = *(const v16bf*)(arow + k + half * 16);
    const __bf16* bp = &bsh[buf][0];
#pragma unroll
    for (int j = 0; j < 4; ++j) {
      v16bf bfr = *(const v16bf*)(bp + (j * 16 + ln) * 32 + half * 16);
      acc[j] = __builtin_amdgcn_wmma_f32_16x16x32_bf16(
          false, af, false, bfr, (short)0, acc[j], false, false);
    }
    if (hasNext) wait_async0();
    __syncthreads();                       // all waves done with buf
    buf ^= 1;
  }

  if (MODE == 0) {
    __bf16* C = (__bf16*)Cout;
#pragma unroll
    for (int j = 0; j < 4; ++j)
#pragma unroll
      for (int v = 0; v < 8; ++v)
        C[(size_t)(tmRow + v + 8 * half) * Nc + bn * 64 + j * 16 + ln] =
            f2bf(acc[j][v] * scale);
  } else if (MODE == 1) {
    __bf16* Vt = (__bf16*)Cout;            // [B,NH,HD,SS]; head == bn
    int b = tmRow / SS;
    int s0 = tmRow % SS;
#pragma unroll
    for (int j = 0; j < 4; ++j) {
      v8bf p;
#pragma unroll
      for (int v = 0; v < 8; ++v) p[v] = f2bf(acc[j][v] * scale);
      *(v8bf*)(Vt + ((size_t)(b * NH + bn) * HD + j * 16 + ln) * SS + s0 +
               half * 8) = p;
    }
  } else {
    float* C = (float*)Cout;
#pragma unroll
    for (int j = 0; j < 4; ++j) {
      int col = bn * 64 + j * 16 + ln;
      float bv = bias[col];
#pragma unroll
      for (int v = 0; v < 8; ++v)
        C[(size_t)(tmRow + v + 8 * half) * Nc + col] = acc[j][v] + bv;
    }
  }
}

// --------------------------------------------------------- flash attention (causal)
// Q is pre-scaled by log2(e)/sqrt(HD): softmax runs in base-2 domain (exp2f),
// which is exactly equivalent and removes all per-score multiplies.
// Q,K: bf16 [B*SS, DMODEL].  Vt: bf16 [B,NH,HD,SS].  Ctx: bf16 [B*SS, DMODEL].
__global__ __launch_bounds__(256) void attn_wmma(
    const __bf16* __restrict__ Q, const __bf16* __restrict__ Kmat,
    const __bf16* __restrict__ Vt, __bf16* __restrict__ Ctx) {
  __shared__ __align__(32) __bf16 pbuf[8][16 * 32];   // per-wave P staging
  const int lane = threadIdx.x & 31;
  const int wave = threadIdx.x >> 5;
  const int ln = lane & 15, half = lane >> 4;
  int wid = blockIdx.x * 8 + wave;
  const int qtPerBH = SS / 16;
  int bh = wid / qtPerBH;
  int qt = wid % qtPerBH;
  int b = bh / NH, h = bh % NH;
  int qBase = qt * 16;

  const __bf16* qrow = Q + (size_t)(b * SS + qBase + ln) * DMODEL + h * HD;
  v16bf qf[2];
#pragma unroll
  for (int d = 0; d < 2; ++d)
    qf[d] = *(const v16bf*)(qrow + d * 32 + half * 16);

  const __bf16* kbase = Kmat + (size_t)(b * SS) * DMODEL + h * HD;
  const __bf16* vbase = Vt + (size_t)bh * HD * SS;

  float Mr[8], Lr[8];
  v8f O[4] = {};
#pragma unroll
  for (int v = 0; v < 8; ++v) { Mr[v] = -INFINITY; Lr[v] = 0.f; }

  auto kkstep = [&](int kkb, auto maskedTag) {
    constexpr bool MASKED = decltype(maskedTag)::value;
    v8f sc[2] = {};
#pragma unroll
    for (int t = 0; t < 2; ++t) {
      const __bf16* krow = kbase + (size_t)(kkb + t * 16 + ln) * DMODEL;
#pragma unroll
      for (int d = 0; d < 2; ++d) {
        v16bf kf = *(const v16bf*)(krow + d * 32 + half * 16);
        sc[t] = __builtin_amdgcn_wmma_f32_16x16x32_bf16(
            false, qf[d], false, kf, (short)0, sc[t], false, false);
      }
    }
#pragma unroll
    for (int v = 0; v < 8; ++v) {
      float a0 = sc[0][v];
      float a1 = sc[1][v];
      if (MASKED) {
        int q = qBase + v + 8 * half;
        if (kkb + ln > q)      a0 = -INFINITY;
        if (kkb + 16 + ln > q) a1 = -INFINITY;
      }
      float m = rowmax16(fmaxf(a0, a1));
      // newM is always finite (column 0 is always causally valid), so
      // exp2(-inf - finite) = 0 handles both the first-step and masked cases.
      float newM = fmaxf(Mr[v], m);
      float scl = exp2f(Mr[v] - newM);
      float p0 = exp2f(a0 - newM);
      float p1 = exp2f(a1 - newM);
      float rs = rowsum16(p0 + p1);
      Lr[v] = Lr[v] * scl + rs;
      Mr[v] = newM;
#pragma unroll
      for (int jd = 0; jd < 4; ++jd) O[jd][v] *= scl;
      int mrow = v + 8 * half;
      pbuf[wave][mrow * 32 + ln]      = f2bf(p0);   // D-layout -> LDS
      pbuf[wave][mrow * 32 + 16 + ln] = f2bf(p1);
    }
    // reload P in A-fragment layout (wave-private; DS ops are in-order)
    v16bf pf = *(const v16bf*)&pbuf[wave][ln * 32 + half * 16];
#pragma unroll
    for (int jd = 0; jd < 4; ++jd) {
      v16bf vf = *(const v16bf*)(vbase + (size_t)(jd * 16 + ln) * SS + kkb +
                                 half * 16);
      O[jd] = __builtin_amdgcn_wmma_f32_16x16x32_bf16(
          false, pf, false, vf, (short)0, O[jd], false, false);
    }
  };

  // all kk blocks before the last are fully inside the causal triangle
  const int lastKkb = ((qBase + 15) / 32) * 32;
  for (int kkb = 0; kkb < lastKkb; kkb += 32)
    kkstep(kkb, std::integral_constant<bool, false>{});
  kkstep(lastKkb, std::integral_constant<bool, true>{});

#pragma unroll
  for (int v = 0; v < 8; ++v) {
    float inv = 1.0f / Lr[v];
    size_t row = (size_t)(b * SS + qBase + v + 8 * half);
#pragma unroll
    for (int jd = 0; jd < 4; ++jd)
      Ctx[row * DMODEL + h * HD + jd * 16 + ln] = f2bf(O[jd][v] * inv);
  }
}

// ---------------------------------------------------------------------- launcher
extern "C" void kernel_launch(void* const* d_in, const int* in_sizes, int n_in,
                              void* d_out, int out_size, void* d_ws,
                              size_t ws_size, hipStream_t stream) {
  (void)in_sizes; (void)n_in; (void)out_size; (void)ws_size;
  const float* x  = (const float*)d_in[0];
  const float* wq = (const float*)d_in[1];
  const float* wk = (const float*)d_in[2];
  const float* wv = (const float*)d_in[3];
  const float* wo = (const float*)d_in[4];
  const float* bo = (const float*)d_in[5];
  float* out = (float*)d_out;

  char* ws = (char*)d_ws;
  size_t off = 0;
  __bf16* xb  = (__bf16*)(ws + off); off += (size_t)MROWS * DIN * 2;
  __bf16* wqT = (__bf16*)(ws + off); off += (size_t)DIN * DMODEL * 2;
  __bf16* wkT = (__bf16*)(ws + off); off += (size_t)DIN * DMODEL * 2;
  __bf16* wvT = (__bf16*)(ws + off); off += (size_t)DIN * DMODEL * 2;
  __bf16* woT = (__bf16*)(ws + off); off += (size_t)DMODEL * DMODEL * 2;
  __bf16* qb  = (__bf16*)(ws + off); off += (size_t)MROWS * DMODEL * 2;
  __bf16* kb  = (__bf16*)(ws + off); off += (size_t)MROWS * DMODEL * 2;
  __bf16* vtb = (__bf16*)(ws + off); off += (size_t)MROWS * DMODEL * 2;
  __bf16* ctx = (__bf16*)(ws + off); off += (size_t)MROWS * DMODEL * 2;

  int nX = MROWS * DIN;
  cast_f32_bf16<<<nX / 4 / 256, 256, 0, stream>>>(x, xb, nX);

  dim3 tgrid(DMODEL / 32, DIN / 32);
  transpose_f32_bf16<<<tgrid, 256, 0, stream>>>(wq, wqT, DIN, DMODEL);
  transpose_f32_bf16<<<tgrid, 256, 0, stream>>>(wk, wkT, DIN, DMODEL);
  transpose_f32_bf16<<<tgrid, 256, 0, stream>>>(wv, wvT, DIN, DMODEL);
  transpose_f32_bf16<<<tgrid, 256, 0, stream>>>(wo, woT, DMODEL, DMODEL);

  // fold 1/sqrt(HD) * log2(e) into Q so attention runs in exp2 domain
  const float smScale = 0.125f * 1.4426950408889634f;
  const int gemmBlocks = (MROWS / 128) * (DMODEL / 64);   // 512
  gemm_bf16_wmma<0><<<gemmBlocks, 256, 0, stream>>>(xb, wqT, qb, nullptr,
                                                    smScale, MROWS, DMODEL, DIN);
  gemm_bf16_wmma<0><<<gemmBlocks, 256, 0, stream>>>(xb, wkT, kb, nullptr,
                                                    1.0f, MROWS, DMODEL, DIN);
  gemm_bf16_wmma<1><<<gemmBlocks, 256, 0, stream>>>(xb, wvT, vtb, nullptr,
                                                    1.0f, MROWS, DMODEL, DIN);

  const int attnBlocks = BB * NH * (SS / 16) / 8;          // 512
  attn_wmma<<<attnBlocks, 256, 0, stream>>>(qb, kb, vtb, ctx);

  gemm_bf16_wmma<2><<<gemmBlocks, 256, 0, stream>>>(ctx, woT, out, bo, 1.0f,
                                                    MROWS, DMODEL, DMODEL);
}
